// MPNNModel_73203422593053
// MI455X (gfx1250) — compile-verified, matrix operand order checked by reading, not run on hardware
//
#include <hip/hip_runtime.h>
#include <hip/hip_bf16.h>

// ---------------------------------------------------------------------------
// MPNN forward for MI455X (gfx1250, wave32, WMMA).
// Dense layers: bf16-WMMA GEMM, f32 accumulate. Tiles staged into LDS with
// CDNA5 async global->LDS DMA (ASYNCcnt), double-buffered so the DMA for
// tile t+1 overlaps the WMMAs of tile t. Weights are pre-transposed to [N,K]
// bf16 once per launch so both A and B fragments are contiguous 16B runs per
// lane (ds_load_b128). Edge MLP is factored through nodes (e == 0 always).
// ---------------------------------------------------------------------------

typedef __attribute__((ext_vector_type(16))) __bf16 v16bf;
typedef __attribute__((ext_vector_type(8)))  float          v8f;
typedef __attribute__((ext_vector_type(8)))  unsigned short us8;

#define NNODES   20000
#define NEDGES   160000
#define INSZ     128
#define HID      256
#define OUTSZ    32
#define NGRAPH   16
#define NUMMP    4
#define NEGSLOPE 0.01f

__device__ __forceinline__ unsigned short f32_to_bf16(float f) {
  unsigned u = __float_as_uint(f);
  u += 0x7fffu + ((u >> 16) & 1u);          // round-to-nearest-even
  return (unsigned short)(u >> 16);
}

// ---------------- elementwise fp32 -> bf16 ----------------
__global__ void k_cvt_bf16(const float* __restrict__ in,
                           unsigned short* __restrict__ out, int n) {
  int i = blockIdx.x * blockDim.x + threadIdx.x;
  int stride = gridDim.x * blockDim.x;
  for (; i < n; i += stride) out[i] = f32_to_bf16(in[i]);
}

// ---------------- transpose + convert: in[K,N] f32 -> out[N,K] bf16 --------
__global__ void k_cvt_bf16_t(const float* __restrict__ in,
                             unsigned short* __restrict__ out, int K, int N) {
  int i = blockIdx.x * blockDim.x + threadIdx.x;
  int stride = gridDim.x * blockDim.x;
  int total = K * N;
  for (; i < total; i += stride) {
    int k = i / N, n = i - k * N;
    out[(size_t)n * K + k] = f32_to_bf16(in[i]);
  }
}

// ---------------- WMMA GEMM: C = act(A[M,K](bf16) @ Wt[N,K](bf16)^T + b) ---
// 128x128 block, 8 waves (4 along M x 2 along N), each wave 32x64 = 2x4 WMMA
// accumulators. K step = 32 (= WMMA K for bf16). LDS double-buffered; tiles
// filled by global_load_async_to_lds_b128 (ASYNCcnt) overlapping the WMMAs.
#define BM 128
#define BN 128
#define BK 32
#define LPAD 8    // 40-half row stride = 80B (multiple of 16B -> aligned b128)

__global__ __launch_bounds__(256)
void k_gemm_bf16(const unsigned short* __restrict__ A,    // [M,K]
                 const unsigned short* __restrict__ Wt,   // [N,K] (W transposed)
                 const float* __restrict__ bias,          // [N] or null
                 float* __restrict__ outF,                // [M,N] or null
                 unsigned short* __restrict__ outB,       // [M,N] or null
                 int M, int N, int K, int act)
{
  __shared__ __align__(16) unsigned short lA[2][BM][BK + LPAD];
  __shared__ __align__(16) unsigned short lB[2][BN][BK + LPAD];

  const int tid  = threadIdx.x;
  const int lane = tid & 31;
  const int wave = tid >> 5;
  const int wm   = wave & 3;             // wave row (M)
  const int wn   = wave >> 2;            // wave col (N)
  const int l15  = lane & 15;
  const int kOff = (lane >> 4) << 3;     // 0 or 8 (lane-half K offset, per ISA layout)
  const int m0   = blockIdx.y * BM;
  const int n0   = blockIdx.x * BN;

  // Staging: 512 16B chunks per tile, 2 per thread (c = tid, tid+256).
  // OOB rows/cols are clamped (they only feed OOB C entries, guarded at store).
  const int rowS0 = tid >> 2;            // chunk row for c = tid
  const int rowS1 = (tid + 256) >> 2;    // chunk row for c = tid + 256
  const int colS  = (tid & 3) << 3;      // k-offset within tile (8 halves)
  int grA0 = m0 + rowS0; grA0 = grA0 < M ? grA0 : M - 1;
  int grA1 = m0 + rowS1; grA1 = grA1 < M ? grA1 : M - 1;
  int gnB0 = n0 + rowS0; gnB0 = gnB0 < N ? gnB0 : N - 1;
  int gnB1 = n0 + rowS1; gnB1 = gnB1 < N ? gnB1 : N - 1;
  const unsigned short* baseA0 = A  + (size_t)grA0 * K + colS;
  const unsigned short* baseA1 = A  + (size_t)grA1 * K + colS;
  const unsigned short* baseB0 = Wt + (size_t)gnB0 * K + colS;
  const unsigned short* baseB1 = Wt + (size_t)gnB1 * K + colS;

  // Async DMA one tile of A and B into LDS buffer `buf` at k-offset k0.
  // VDST = per-lane LDS byte offset (low 32 bits of the generic address).
  auto stage = [&](int buf, int k0) {
    unsigned la0 = (unsigned)(size_t)&lA[buf][rowS0][colS];
    unsigned la1 = (unsigned)(size_t)&lA[buf][rowS1][colS];
    unsigned lb0 = (unsigned)(size_t)&lB[buf][rowS0][colS];
    unsigned lb1 = (unsigned)(size_t)&lB[buf][rowS1][colS];
    asm volatile("global_load_async_to_lds_b128 %0, %1, off"
                 :: "v"(la0), "v"((unsigned long long)(size_t)(baseA0 + k0)) : "memory");
    asm volatile("global_load_async_to_lds_b128 %0, %1, off"
                 :: "v"(la1), "v"((unsigned long long)(size_t)(baseA1 + k0)) : "memory");
    asm volatile("global_load_async_to_lds_b128 %0, %1, off"
                 :: "v"(lb0), "v"((unsigned long long)(size_t)(baseB0 + k0)) : "memory");
    asm volatile("global_load_async_to_lds_b128 %0, %1, off"
                 :: "v"(lb1), "v"((unsigned long long)(size_t)(baseB1 + k0)) : "memory");
    // Speculative prefetch one more tile ahead (OOB prefetches are dropped).
    __builtin_prefetch(baseA0 + k0 + BK, 0, 3);      // global_prefetch_b8
    __builtin_prefetch(baseA1 + k0 + BK, 0, 3);
    __builtin_prefetch(baseB0 + k0 + BK, 0, 3);
    __builtin_prefetch(baseB1 + k0 + BK, 0, 3);
  };

  v8f acc[2][4];
  #pragma unroll
  for (int fm = 0; fm < 2; ++fm)
    #pragma unroll
    for (int fn = 0; fn < 4; ++fn)
      acc[fm][fn] = {};

  const int nk = K / BK;                 // K is always a multiple of 32 here
  stage(0, 0);
  asm volatile("s_wait_asynccnt 0x0" ::: "memory");
  __syncthreads();

  for (int kt = 0; kt < nk; ++kt) {
    const int buf = kt & 1;
    if (kt + 1 < nk) stage(buf ^ 1, (kt + 1) * BK);   // DMA next tile during WMMAs

    // Fragment load per CDNA5 16-bit layouts. Per lane both A and B fragments
    // are two contiguous 16B runs: k = kOff+0..7 (elems 0..7) and
    // k = kOff+16..23 (elems 8..15)  -> 2x ds_load_b128 each.
    union F { v16bf v; unsigned short u[16]; };
    F fa[2], fb[4];
    #pragma unroll
    for (int e = 0; e < 16; ++e) {
      int j  = e >> 1;
      int kk = ((j < 4) ? (j << 1) : ((j << 1) + 8)) + kOff + (e & 1);
      #pragma unroll
      for (int fm = 0; fm < 2; ++fm)
        fa[fm].u[e] = lA[buf][wm * 32 + fm * 16 + l15][kk];
      #pragma unroll
      for (int fn = 0; fn < 4; ++fn)
        fb[fn].u[e] = lB[buf][wn * 64 + fn * 16 + l15][kk];
    }

    #pragma unroll
    for (int fm = 0; fm < 2; ++fm)
      #pragma unroll
      for (int fn = 0; fn < 4; ++fn)
        acc[fm][fn] = __builtin_amdgcn_wmma_f32_16x16x32_bf16(
            false, fa[fm].v, false, fb[fn].v, (short)0, acc[fm][fn],
            false, false);

    // Wave's next-tile DMA must have landed, and all waves must be done
    // reading this buffer (it is re-staged next iteration).
    asm volatile("s_wait_asynccnt 0x0" ::: "memory");
    __syncthreads();
  }

  // Epilogue: C/D layout -> lanes 0-15: N=lane, M=r; lanes 16-31: N=lane-16, M=r+8.
  #pragma unroll
  for (int fm = 0; fm < 2; ++fm) {
    int mBase = m0 + wm * 32 + fm * 16 + ((lane >> 4) << 3);
    #pragma unroll
    for (int fn = 0; fn < 4; ++fn) {
      int gn = n0 + wn * 64 + fn * 16 + l15;
      if (gn >= N) continue;
      float bv = bias ? bias[gn] : 0.0f;
      #pragma unroll
      for (int r = 0; r < 8; ++r) {
        int gm = mBase + r;
        if (gm >= M) continue;
        float v = acc[fm][fn][r] + bv;
        if (act) v = (v >= 0.0f) ? v : NEGSLOPE * v;
        size_t o = (size_t)gm * N + gn;
        if (outF) outF[o] = v;
        if (outB) outB[o] = f32_to_bf16(v);
      }
    }
  }
}

// ---------------- degree (incl. self-loop) ----------------
__global__ void k_deg_init(int* deg, int n) {
  int i = blockIdx.x * blockDim.x + threadIdx.x;
  if (i < n) deg[i] = 1;                         // self loop
}
__global__ void k_deg_scatter(const int* __restrict__ dst, int* __restrict__ deg, int nE) {
  int e = blockIdx.x * blockDim.x + threadIdx.x;
  if (e < nE) atomicAdd(&deg[dst[e]], 1);
}

// agg[i] = deg[i]*(tDst[i]+b) + tSrc[i]   (in-place into tDst)
__global__ void k_agg_init(float* __restrict__ tDst, const float* __restrict__ tSrc,
                           const float* __restrict__ eb, const int* __restrict__ deg) {
  int i = blockIdx.x, c = threadIdx.x;
  size_t o = (size_t)i * HID + c;
  tDst[o] = (float)deg[i] * (tDst[o] + eb[c]) + tSrc[o];
}

// agg[dst[e]] += tSrc[src[e]]  (one block per edge, one lane per channel)
__global__ void k_agg_scatter(const int* __restrict__ src, const int* __restrict__ dst,
                              const float* __restrict__ tSrc, float* __restrict__ agg) {
  int e = blockIdx.x, c = threadIdx.x;
  int s = src[e], d = dst[e];
  float v = tSrc[(size_t)s * HID + c];
  __hip_atomic_fetch_add(&agg[(size_t)d * HID + c], v,
                         __ATOMIC_RELAXED, __HIP_MEMORY_SCOPE_AGENT);
}

// ---------------- pooling ----------------
__global__ void k_pool_init(float* sums, float* maxs, int* cnt) {
  int i = blockIdx.x * blockDim.x + threadIdx.x;
  if (i < NGRAPH * HID) { sums[i] = 0.0f; maxs[i] = -3.402823466e38f; }
  if (i < NGRAPH) cnt[i] = 0;
}
__device__ __forceinline__ void atomicMaxF(float* addr, float value) {
  if (value >= 0.0f) atomicMax((int*)addr, __float_as_int(value));
  else atomicMin((unsigned int*)addr, (unsigned int)__float_as_int(value));
}
__global__ void k_pool_scatter(const float* __restrict__ h, const int* __restrict__ batch,
                               float* __restrict__ sums, float* __restrict__ maxs,
                               int* __restrict__ cnt) {
  int i = blockIdx.x, c = threadIdx.x;
  int g = batch[i];
  float v = h[(size_t)i * HID + c];
  __hip_atomic_fetch_add(&sums[g * HID + c], v, __ATOMIC_RELAXED, __HIP_MEMORY_SCOPE_AGENT);
  atomicMaxF(&maxs[g * HID + c], v);
  if (c == 0) atomicAdd(&cnt[g], 1);
}
__global__ void k_pool_fin(const float* __restrict__ sums, const float* __restrict__ maxs,
                           const int* __restrict__ cnt, float* __restrict__ pooled) {
  int i = blockIdx.x * blockDim.x + threadIdx.x;
  if (i >= NGRAPH * HID) return;
  int g = i >> 8, c = i & 255;
  float s = sums[i];
  pooled[g * 3 * HID + c]           = s / fmaxf((float)cnt[g], 1.0f);
  pooled[g * 3 * HID + HID + c]     = maxs[i];
  pooled[g * 3 * HID + 2 * HID + c] = s;
}

// ---------------------------------------------------------------------------
static inline void gemm(hipStream_t s, const unsigned short* A, const unsigned short* Wt,
                        const float* bias, float* outF, unsigned short* outB,
                        int M, int N, int K, int act) {
  dim3 g((N + BN - 1) / BN, (M + BM - 1) / BM);
  k_gemm_bf16<<<g, 256, 0, s>>>(A, Wt, bias, outF, outB, M, N, K, act);
}

extern "C" void kernel_launch(void* const* d_in, const int* in_sizes, int n_in,
                              void* d_out, int out_size, void* d_ws, size_t ws_size,
                              hipStream_t stream) {
  (void)in_sizes; (void)n_in; (void)out_size; (void)ws_size;

  const float* x         = (const float*)d_in[0];
  const int*   ei        = (const int*)d_in[1];      // [2, E]: src row then dst row
  const int*   batch     = (const int*)d_in[2];
  const float* enc_Win   = (const float*)d_in[3];
  const float* enc_bin   = (const float*)d_in[4];
  const float* enc_Wh    = (const float*)d_in[5];
  const float* enc_bh    = (const float*)d_in[6];
  const float* enc_Wout  = (const float*)d_in[7];
  const float* enc_bout  = (const float*)d_in[8];
  const float* edge_W    = (const float*)d_in[9];
  const float* edge_b    = (const float*)d_in[10];
  const float* proc_Win  = (const float*)d_in[11];
  const float* proc_bin  = (const float*)d_in[12];
  const float* proc_Wh   = (const float*)d_in[13];
  const float* proc_bh   = (const float*)d_in[14];
  const float* proc_Wout = (const float*)d_in[15];
  const float* proc_bout = (const float*)d_in[16];
  const float* dec_Win   = (const float*)d_in[17];
  const float* dec_bin   = (const float*)d_in[18];
  const float* dec_Wh    = (const float*)d_in[19];
  const float* dec_bh    = (const float*)d_in[20];
  const float* dec_Wout  = (const float*)d_in[21];
  const float* dec_bout  = (const float*)d_in[22];

  const int* src = ei;
  const int* dst = ei + NEDGES;

  // ---- workspace carve ----
  char* ws = (char*)d_ws;
  size_t off = 0;
  auto carve = [&](size_t bytes) -> char* {
    char* p = ws + off;
    off += (bytes + 255) & ~(size_t)255;
    return p;
  };
  // transposed bf16 weights [N,K]
  unsigned short* t_enc_Win  = (unsigned short*)carve((size_t)INSZ * HID * 2);
  unsigned short* t_enc_Wh0  = (unsigned short*)carve((size_t)HID * HID * 2);
  unsigned short* t_enc_Wh1  = (unsigned short*)carve((size_t)HID * HID * 2);
  unsigned short* t_enc_Wout = (unsigned short*)carve((size_t)HID * HID * 2);
  unsigned short* t_edge_Wd  = (unsigned short*)carve((size_t)HID * HID * 2);  // rows 0:256
  unsigned short* t_edge_Ws  = (unsigned short*)carve((size_t)HID * HID * 2);  // rows 256:512
  unsigned short* t_proc_Win = (unsigned short*)carve((size_t)HID * HID * 2);
  unsigned short* t_proc_Wh0 = (unsigned short*)carve((size_t)HID * HID * 2);
  unsigned short* t_proc_Wh1 = (unsigned short*)carve((size_t)HID * HID * 2);
  unsigned short* t_proc_Wout= (unsigned short*)carve((size_t)HID * HID * 2);
  unsigned short* t_dec_Win  = (unsigned short*)carve((size_t)3 * HID * HID * 2);
  unsigned short* t_dec_Wh0  = (unsigned short*)carve((size_t)HID * HID * 2);
  unsigned short* t_dec_Wh1  = (unsigned short*)carve((size_t)HID * HID * 2);
  unsigned short* t_dec_Wout = (unsigned short*)carve((size_t)HID * OUTSZ * 2);

  unsigned short* xB   = (unsigned short*)carve((size_t)NNODES * INSZ * 2);
  unsigned short* mB1  = (unsigned short*)carve((size_t)NNODES * HID * 2);
  unsigned short* mB2  = (unsigned short*)carve((size_t)NNODES * HID * 2);
  unsigned short* hB   = (unsigned short*)carve((size_t)NNODES * HID * 2);
  unsigned short* aggB = (unsigned short*)carve((size_t)NNODES * HID * 2);
  float* hF   = (float*)carve((size_t)NNODES * HID * 4);
  float* tDst = (float*)carve((size_t)NNODES * HID * 4);   // becomes agg in-place
  float* tSrc = (float*)carve((size_t)NNODES * HID * 4);
  int*   deg  = (int*)carve((size_t)NNODES * 4);
  float* sums = (float*)carve((size_t)NGRAPH * HID * 4);
  float* maxs = (float*)carve((size_t)NGRAPH * HID * 4);
  int*   cnt  = (int*)carve((size_t)NGRAPH * 4);
  float* pooled = (float*)carve((size_t)NGRAPH * 3 * HID * 4);
  unsigned short* pooledB = (unsigned short*)carve((size_t)NGRAPH * 3 * HID * 2);

  auto cvt = [&](const float* in, unsigned short* out, int n) {
    int blocks = (n + 255) / 256; if (blocks > 4096) blocks = 4096;
    k_cvt_bf16<<<blocks, 256, 0, stream>>>(in, out, n);
  };
  auto cvtT = [&](const float* in, unsigned short* out, int K, int N) {
    int n = K * N;
    int blocks = (n + 255) / 256; if (blocks > 4096) blocks = 4096;
    k_cvt_bf16_t<<<blocks, 256, 0, stream>>>(in, out, K, N);
  };

  // ---- weight transpose+convert (once per launch; ~1M elements total) ----
  cvtT(enc_Win,              t_enc_Win,  INSZ, HID);
  cvtT(enc_Wh,               t_enc_Wh0,  HID,  HID);
  cvtT(enc_Wh + HID * HID,   t_enc_Wh1,  HID,  HID);
  cvtT(enc_Wout,             t_enc_Wout, HID,  HID);
  cvtT(edge_W,               t_edge_Wd,  HID,  HID);   // edge_W rows 0:256
  cvtT(edge_W + HID * HID,   t_edge_Ws,  HID,  HID);   // edge_W rows 256:512
  cvtT(proc_Win,             t_proc_Win, HID,  HID);
  cvtT(proc_Wh,              t_proc_Wh0, HID,  HID);
  cvtT(proc_Wh + HID * HID,  t_proc_Wh1, HID,  HID);
  cvtT(proc_Wout,            t_proc_Wout,HID,  HID);
  cvtT(dec_Win,              t_dec_Win,  3 * HID, HID);
  cvtT(dec_Wh,               t_dec_Wh0,  HID,  HID);
  cvtT(dec_Wh + HID * HID,   t_dec_Wh1,  HID,  HID);
  cvtT(dec_Wout,             t_dec_Wout, HID,  OUTSZ);
  cvt(x, xB, NNODES * INSZ);

  // ---- encoder MLP ----
  gemm(stream, xB,  t_enc_Win,  enc_bin,      nullptr, mB1, NNODES, HID, INSZ, 1);
  gemm(stream, mB1, t_enc_Wh0,  enc_bh,       nullptr, mB2, NNODES, HID, HID, 1);
  gemm(stream, mB2, t_enc_Wh1,  enc_bh + HID, nullptr, mB1, NNODES, HID, HID, 1);
  gemm(stream, mB1, t_enc_Wout, enc_bout,     nullptr, hB,  NNODES, HID, HID, 0);

  // ---- degree (constant across rounds) ----
  k_deg_init<<<(NNODES + 255) / 256, 256, 0, stream>>>(deg, NNODES);
  k_deg_scatter<<<(NEDGES + 255) / 256, 256, 0, stream>>>(dst, deg, NEDGES);

  // ---- 4 message-passing rounds (shared weights) ----
  for (int r = 0; r < NUMMP; ++r) {
    // edge MLP factored through nodes: tDst = h@W[0:256], tSrc = h@W[256:512]
    gemm(stream, hB, t_edge_Wd, nullptr, tDst, nullptr, NNODES, HID, HID, 0);
    gemm(stream, hB, t_edge_Ws, nullptr, tSrc, nullptr, NNODES, HID, HID, 0);
    k_agg_init<<<NNODES, HID, 0, stream>>>(tDst, tSrc, edge_b, deg);
    k_agg_scatter<<<NEDGES, HID, 0, stream>>>(src, dst, tSrc, tDst);
    cvt(tDst, aggB, NNODES * HID);
    // processor MLP
    gemm(stream, aggB, t_proc_Win,  proc_bin,      nullptr, mB1, NNODES, HID, HID, 1);
    gemm(stream, mB1,  t_proc_Wh0,  proc_bh,       nullptr, mB2, NNODES, HID, HID, 1);
    gemm(stream, mB2,  t_proc_Wh1,  proc_bh + HID, nullptr, mB1, NNODES, HID, HID, 1);
    gemm(stream, mB1,  t_proc_Wout, proc_bout,     hF,      hB,  NNODES, HID, HID, 0);
  }

  // ---- pooling (mean | max | sum) ----
  k_pool_init<<<(NGRAPH * HID + 255) / 256, 256, 0, stream>>>(sums, maxs, cnt);
  k_pool_scatter<<<NNODES, HID, 0, stream>>>(hF, batch, sums, maxs, cnt);
  k_pool_fin<<<(NGRAPH * HID + 255) / 256, 256, 0, stream>>>(sums, maxs, cnt, pooled);
  cvt(pooled, pooledB, NGRAPH * 3 * HID);

  // ---- decoder MLP ----
  gemm(stream, pooledB, t_dec_Win,  dec_bin,      nullptr, mB1, NGRAPH, HID,   3 * HID, 1);
  gemm(stream, mB1,     t_dec_Wh0,  dec_bh,       nullptr, mB2, NGRAPH, HID,   HID,     1);
  gemm(stream, mB2,     t_dec_Wh1,  dec_bh + HID, nullptr, mB1, NGRAPH, HID,   HID,     1);
  gemm(stream, mB1,     t_dec_Wout, dec_bout, (float*)d_out, nullptr, NGRAPH, OUTSZ, HID, 0);
}
